// STRNN_34677565948678
// MI455X (gfx1250) — matrix operand order repaired and checked by reading, not compiled, not run on hardware
//
#include <hip/hip_runtime.h>

typedef __attribute__((ext_vector_type(16))) _Float16 v16h;
typedef __attribute__((ext_vector_type(8)))  _Float16 v8h;
typedef __attribute__((ext_vector_type(8)))  float    v8f;

#define B_   128
#define T_   16
#define E_   62
#define C_   5
#define H_   128
#define SPP_ 16
#define TPP_ 8
#define NC_  3
#define D_   4

// workspace layout (bytes)
#define WS_WFRAG   0u          // 8 mats * 4 kc * 8 nt * 32 * 16 halves * 2B = 262144
#define WS_UFRAG   262144u     // 4 dirs * 8 nt * 32 * 16 halves * 2B       = 32768
#define WS_NBR     294912u     // D*E*4 ints  (pad 4096)
#define WS_NBRW    299008u     // D*E*4 floats (pad 4096)
#define WS_WPROJ   303104u     // D*E floats (pad 1024)
#define WS_MBIAS   304128u     // 1 float (pad 64)
#define WS_M4      304192u     // D*T*B*H floats = 4194304
#define WS_M16     4498496u    // T*B*H halves   = 524288
#define WS_RWS     5022784u    // 2*T*B floats   = 16384
// total ~4.8 MB

// ---------------------------------------------------------------------------
// Prep 1: reformat the 8 recurrent 128x128 matrices (4x W_w, f_rs, f_vs,
// b_rs, b_vs) AND the 4 U_w (128x5, K-padded to 32) into the exact
// v_wmma_f32_16x16x32_f16 B-operand lane layout:
//   frag[..][lane][e] = M[n][k], n = nt*16 + lane%16, k = kc*32 + (e%8) + 8*(lane/16) + 16*(e/8)
// ---------------------------------------------------------------------------
__global__ __launch_bounds__(256) void stg_prep_frag(
    const float* __restrict__ Ww,  const float* __restrict__ frs,
    const float* __restrict__ fvs, const float* __restrict__ brs,
    const float* __restrict__ bvs, const float* __restrict__ Uw,
    _Float16* __restrict__ wfrag,  _Float16* __restrict__ ufrag)
{
  int g = blockIdx.x * 256 + threadIdx.x;      // [0, 131072 + 16384)
  if (g < 131072) {
    int mat  = g >> 14;
    int pos  = g & 16383;
    int kc   = pos >> 12;
    int nt   = (pos >> 9) & 7;
    int lane = (pos >> 4) & 31;
    int e    = pos & 15;
    int k = kc * 32 + (e & 7) + ((lane >> 4) << 3) + ((e >> 3) << 4);
    int n = nt * 16 + (lane & 15);
    float v;
    if (mat < 4)       v = Ww[(mat * H_ + n) * H_ + k];
    else if (mat == 4) v = frs[n * H_ + k];
    else if (mat == 5) v = fvs[n * H_ + k];
    else if (mat == 6) v = brs[n * H_ + k];
    else               v = bvs[n * H_ + k];
    wfrag[g] = (_Float16)v;
  } else {
    int gu   = g - 131072;
    int d    = gu >> 12;
    int pos  = gu & 4095;
    int nt   = pos >> 9;
    int lane = (pos >> 4) & 31;
    int e    = pos & 15;
    int k = (e & 7) + ((lane >> 4) << 3) + ((e >> 3) << 4);            // K-chunk 0 only (K=32, 5 valid)
    int n = nt * 16 + (lane & 15);
    float v = (k < C_) ? Uw[(d * H_ + n) * C_ + k] : 0.f;
    ufrag[gu] = (_Float16)v;
  }
}

// ---------------------------------------------------------------------------
// Prep 2: compact the sparse Ns rows to <=3 neighbor (index,weight) pairs,
// collapse P_w/ps_w to one scalar per (d, step), fold P_b/ps_b into one bias.
// ---------------------------------------------------------------------------
__global__ void stg_prep_graph(
    const float* __restrict__ Ns, const float* __restrict__ Pw,
    const float* __restrict__ Pb, const float* __restrict__ psw,
    const float* __restrict__ psb,
    int* __restrict__ nbr, float* __restrict__ nbrw,
    float* __restrict__ wproj, float* __restrict__ mbias)
{
  int tid = threadIdx.x;
  if (tid < D_ * E_) {
    int d = tid / E_, pi = tid % E_;
    float wp = 0.f;
    for (int p = 0; p < SPP_; ++p)
      wp += psw[d * SPP_ + p] * Pw[(d * SPP_ + p) * E_ + pi];
    wproj[tid] = wp;
    int cnt = 0; int idx[3] = {0, 0, 0}; float w[3] = {0.f, 0.f, 0.f};
    const float* row = Ns + (d * E_ + pi) * E_;
    for (int e = 0; e < E_; ++e) {
      float v = row[e];
      if (v != 0.f && cnt < 3) { idx[cnt] = e; w[cnt] = v; ++cnt; }
    }
    nbr[tid * 4 + 0] = idx[0]; nbr[tid * 4 + 1] = idx[1];
    nbr[tid * 4 + 2] = idx[2]; nbr[tid * 4 + 3] = cnt;
    nbrw[tid * 4 + 0] = w[0];  nbrw[tid * 4 + 1] = w[1];
    nbrw[tid * 4 + 2] = w[2];  nbrw[tid * 4 + 3] = 0.f;
  }
  if (tid == 255) {
    float mb = psb[0];
    for (int d = 0; d < D_; ++d)
      for (int p = 0; p < SPP_; ++p)
        mb += psw[d * SPP_ + p] * Pb[d * SPP_ + p];
    mbias[0] = mb;
  }
}

// ---------------------------------------------------------------------------
// Spatial RNN chain: one workgroup per (d, t, batch-tile-of-16). The full
// 62-step history lives in LDS (f16); W and U fragments stay in VGPRs.
// Per step each of the 8 waves runs 5 chained v_wmma_f32_16x16x32_f16:
// 1 for the K=32-padded input term (x@U.T) + 4 for the K=128 recurrence.
// Neighbor aggregation is packed-f16 VALU (<=3 v8h fma per thread).
// ---------------------------------------------------------------------------
__global__ __launch_bounds__(256) void stg_chain(
    const float* __restrict__ x,    const int* __restrict__ perm,
    const _Float16* __restrict__ wfrag, const _Float16* __restrict__ ufrag,
    const float* __restrict__ Ub,   const float* __restrict__ Wb,
    const float* __restrict__ spbs, const int* __restrict__ nbr,
    const float* __restrict__ nbrw, const float* __restrict__ wproj,
    float* __restrict__ m4)
{
  __shared__ _Float16 hist[E_][16][128];   // 253952 B : state history (f16)
  __shared__ _Float16 hagg[16][128];       //   4096 B : neighbor aggregate (WMMA A)
  __shared__ _Float16 xq[E_][16][32];      //  63488 B : perm-gathered x, K-padded f16
  __shared__ int      snb[E_][4];
  __shared__ float    snw[E_][4];          // w0,w1,w2, wproj

  int wg    = blockIdx.x;
  int btile = wg & 7;
  int t     = (wg >> 3) & 15;
  int d     = wg >> 7;
  int tid   = threadIdx.x;
  int lane  = tid & 31;
  int nt    = tid >> 5;
  int ln15  = lane & 15;
  int hi    = lane >> 4;
  int gn    = nt * 16 + ln15;              // this lane's output hidden unit

  for (int i = tid; i < E_; i += 256) {
    int base = (d * E_ + i) * 4;
    snb[i][0] = nbr[base];     snb[i][1] = nbr[base + 1];
    snb[i][2] = nbr[base + 2]; snb[i][3] = nbr[base + 3];
    snw[i][0] = nbrw[base];    snw[i][1] = nbrw[base + 1];
    snw[i][2] = nbrw[base + 2];
    snw[i][3] = wproj[d * E_ + i];
  }
  // zero-pad x fragment then fill the 5 valid K entries
  for (int i = tid; i < E_ * 16 * 32 / 8; i += 256)
    ((v8h*)xq)[i] = (v8h)(_Float16)0.f;
  __syncthreads();
  for (int i = tid; i < E_ * 16 * C_; i += 256) {
    int pi = i / (16 * C_); int rem = i % (16 * C_);
    int m  = rem / C_;      int c   = rem % C_;
    int el = perm[d * E_ + pi];
    int b  = btile * 16 + m;
    xq[pi][m][c] = (_Float16)x[((b * T_ + t) * E_ + el) * C_ + c];
  }

  // resident weight fragments: 4 K-chunks of W (B-operand) + 1 of U
  v16h bw[4];
#pragma unroll
  for (int kc = 0; kc < 4; ++kc)
    bw[kc] = *(const v16h*)(wfrag + (((unsigned)(d * 4 + kc) * 8 + nt) * 32 + lane) * 16);
  v16h bu = *(const v16h*)(ufrag + (((unsigned)d * 8 + nt) * 32 + lane) * 16);

  float biasn = Ub[d * H_ + gn] + Wb[d * H_ + gn] + spbs[d * H_ + gn];
  float macc[8];
#pragma unroll
  for (int r = 0; r < 8; ++r) macc[r] = 0.f;

  __syncthreads();

  for (int pi = 0; pi < E_; ++pi) {
    // ---- h_agg: <=3 packed-f16 vector FMAs per thread ----
    {
      int cnt  = snb[pi][3];
      int base = tid * 8;
      int m    = base >> 7;
      int k0   = base & 127;
      v8h s = (v8h)(_Float16)0.f;
      for (int q = 0; q < cnt; ++q) {
        int      e = snb[pi][q];
        _Float16 w = (_Float16)snw[pi][q];
        v8h hv = *(const v8h*)(&hist[e][m][k0]);
        s += hv * w;
      }
      *(v8h*)(&hagg[m][k0]) = s;
    }
    __syncthreads();

    // ---- 5 chained WMMAs: x@U.T (K=32 padded) + h_agg@W.T (K=128) ----
    v8f acc = {};
    {
      int kb = hi ? 8 : 0;
      v8h lo = *(const v8h*)(&xq[pi][ln15][kb]);
      v8h hv = *(const v8h*)(&xq[pi][ln15][kb + 16]);
      v16h a = __builtin_shufflevector(lo, hv, 0, 1, 2, 3, 4, 5, 6, 7,
                                       8, 9, 10, 11, 12, 13, 14, 15);
      acc = __builtin_amdgcn_wmma_f32_16x16x32_f16(
          false, a, false, bu, (short)0, acc, false, false);
    }
#pragma unroll
    for (int kc = 0; kc < 4; ++kc) {
      int kb = kc * 32 + (hi ? 8 : 0);
      v8h lo = *(const v8h*)(&hagg[ln15][kb]);
      v8h hv = *(const v8h*)(&hagg[ln15][kb + 16]);
      v16h a = __builtin_shufflevector(lo, hv, 0, 1, 2, 3, 4, 5, 6, 7,
                                       8, 9, 10, 11, 12, 13, 14, 15);
      acc = __builtin_amdgcn_wmma_f32_16x16x32_f16(
          false, a, false, bw[kc], (short)0, acc, false, false);
    }

    // ---- epilogue: bias, relu, store state, accumulate projection ----
    float wp = snw[pi][3];
#pragma unroll
    for (int r = 0; r < 8; ++r) {
      int m = r + hi * 8;
      float v = acc[r] + biasn;
      v = v > 0.f ? v : 0.f;
      hist[pi][m][gn] = (_Float16)v;
      macc[r] += wp * v;
    }
    __syncthreads();   // hist writes visible + hagg free before next iteration
  }

#pragma unroll
  for (int r = 0; r < 8; ++r) {
    int b = btile * 16 + r + hi * 8;
    m4[((d * T_ + t) * B_ + b) * H_ + gn] = macc[r];
  }
}

// ---------------------------------------------------------------------------
// Reduce the 4 directional partials into the temporal-RNN input (f16).
// ---------------------------------------------------------------------------
__global__ __launch_bounds__(256) void stg_reduce_m(
    const float* __restrict__ m4, const float* __restrict__ mbias,
    _Float16* __restrict__ m16)
{
  int g = blockIdx.x * 256 + threadIdx.x;    // T*B*H = 262144
  float s = mbias[0];
#pragma unroll
  for (int d = 0; d < D_; ++d) s += m4[d * (T_ * B_ * H_) + g];
  m16[g] = (_Float16)s;
}

// ---------------------------------------------------------------------------
// Temporal RNN (fwd + bwd): one workgroup per (dir, batch-tile); each of 16
// sequential steps is 8 chained WMMAs (m_t@rs.T + h@vs.T); the fp1 head
// collapses to a per-(step,row) scalar reduced with shuffles + LDS atomics.
// ---------------------------------------------------------------------------
__global__ __launch_bounds__(256) void stg_trnn(
    const _Float16* __restrict__ m16, const _Float16* __restrict__ wfrag,
    const float* __restrict__ frsb, const float* __restrict__ fvsb,
    const float* __restrict__ fbs,  const float* __restrict__ fp1w,
    const float* __restrict__ brsb, const float* __restrict__ bvsb,
    const float* __restrict__ bbs,  const float* __restrict__ bp1w,
    float* __restrict__ rws)
{
  __shared__ _Float16 hprev[16][128];
  __shared__ float    rsum[T_][16];

  int wg    = blockIdx.x;
  int dir   = wg >> 3;
  int btile = wg & 7;
  int tid   = threadIdx.x;
  int lane  = tid & 31;
  int nt    = tid >> 5;
  int ln15  = lane & 15;
  int hi    = lane >> 4;
  int gn    = nt * 16 + ln15;

  for (int i = tid; i < 16 * 128; i += 256) ((_Float16*)hprev)[i] = (_Float16)0.f;
  for (int i = tid; i < T_ * 16; i += 256) ((float*)rsum)[i] = 0.f;

  int matr = 4 + dir * 2, matv = 5 + dir * 2;
  v16h br[4], bv[4];
#pragma unroll
  for (int kc = 0; kc < 4; ++kc) {
    br[kc] = *(const v16h*)(wfrag + (((unsigned)(matr * 4 + kc) * 8 + nt) * 32 + lane) * 16);
    bv[kc] = *(const v16h*)(wfrag + (((unsigned)(matv * 4 + kc) * 8 + nt) * 32 + lane) * 16);
  }
  float biasn = dir == 0 ? frsb[gn] + fvsb[gn] + fbs[gn]
                         : brsb[gn] + bvsb[gn] + bbs[gn];
  float p1 = dir == 0 ? fp1w[gn] : bp1w[gn];

  __syncthreads();

  for (int s = 0; s < T_; ++s) {
    int tm = dir == 0 ? s : (T_ - 1 - s);
    const _Float16* mrow = m16 + ((unsigned)(tm * B_ + btile * 16 + ln15)) * H_;
    v8f acc = {};
#pragma unroll
    for (int kc = 0; kc < 4; ++kc) {                 // m_t @ rs_w.T
      int kb = kc * 32 + (hi ? 8 : 0);
      v8h lo = *(const v8h*)(mrow + kb);
      v8h hv = *(const v8h*)(mrow + kb + 16);
      v16h a = __builtin_shufflevector(lo, hv, 0, 1, 2, 3, 4, 5, 6, 7,
                                       8, 9, 10, 11, 12, 13, 14, 15);
      acc = __builtin_amdgcn_wmma_f32_16x16x32_f16(
          false, a, false, br[kc], (short)0, acc, false, false);
    }
#pragma unroll
    for (int kc = 0; kc < 4; ++kc) {                 // + h_prev @ vs_w.T
      int kb = kc * 32 + (hi ? 8 : 0);
      v8h lo = *(const v8h*)(&hprev[ln15][kb]);
      v8h hv = *(const v8h*)(&hprev[ln15][kb + 16]);
      v16h a = __builtin_shufflevector(lo, hv, 0, 1, 2, 3, 4, 5, 6, 7,
                                       8, 9, 10, 11, 12, 13, 14, 15);
      acc = __builtin_amdgcn_wmma_f32_16x16x32_f16(
          false, a, false, bv[kc], (short)0, acc, false, false);
    }
    __syncthreads();   // all hprev reads complete before overwrite

#pragma unroll
    for (int r = 0; r < 8; ++r) {
      int m = r + hi * 8;
      float v = acc[r] + biasn;
      v = v > 0.f ? v : 0.f;
      hprev[m][gn] = (_Float16)v;
      float sr = v * p1;                 // head projection contribution
      sr += __shfl_xor(sr, 1, 32);
      sr += __shfl_xor(sr, 2, 32);
      sr += __shfl_xor(sr, 4, 32);
      sr += __shfl_xor(sr, 8, 32);
      if (ln15 == 0) atomicAdd(&rsum[s][m], sr);
    }
    __syncthreads();
  }

  for (int i = tid; i < T_ * 16; i += 256) {
    int s = i >> 4, m = i & 15;
    rws[(dir * T_ + s) * B_ + btile * 16 + m] = rsum[s][m];
  }
}

// ---------------------------------------------------------------------------
// Head: zf/zb and the final class projection (tiny).
// ---------------------------------------------------------------------------
__global__ void stg_final(
    const float* __restrict__ rws,
    const float* __restrict__ fpjw, const float* __restrict__ fpjb,
    const float* __restrict__ fp1w, const float* __restrict__ fp1b,
    const float* __restrict__ fp2w, const float* __restrict__ fp2b,
    const float* __restrict__ bpjw, const float* __restrict__ bpjb,
    const float* __restrict__ bp1w, const float* __restrict__ bp1b,
    const float* __restrict__ bp2w, const float* __restrict__ bp2b,
    float* __restrict__ out)
{
  int b = threadIdx.x;   // 128
  float Sf = 0.f, Sb = 0.f;
  for (int h = 0; h < H_; ++h) { Sf += fp1w[h]; Sb += bp1w[h]; }
  float rf[T_], rb[T_];
  for (int t = 0; t < T_; ++t) {
    rf[t] = rws[t * B_ + b];
    rb[t] = rws[(T_ + t) * B_ + b];
  }
  float o[NC_];
  for (int nc = 0; nc < NC_; ++nc) o[nc] = fp2b[nc] + bp2b[nc];
  for (int p = 0; p < TPP_; ++p) {
    float zf = fp1b[0] + fpjb[p] * Sf;
    float zb = bp1b[0] + bpjb[p] * Sb;
    for (int t = 0; t < T_; ++t) {
      zf += rf[t] * fpjw[p * T_ + t];
      zb += rb[t] * bpjw[p * T_ + t];
    }
    for (int nc = 0; nc < NC_; ++nc)
      o[nc] += zf * fp2w[nc * TPP_ + p] + zb * bp2w[nc * TPP_ + p];
  }
  for (int nc = 0; nc < NC_; ++nc) out[b * NC_ + nc] = o[nc];
}

// ---------------------------------------------------------------------------
extern "C" void kernel_launch(void* const* d_in, const int* in_sizes, int n_in,
                              void* d_out, int out_size, void* d_ws, size_t ws_size,
                              hipStream_t stream)
{
  (void)in_sizes; (void)n_in; (void)out_size; (void)ws_size;
  const float* x    = (const float*)d_in[0];
  const float* Ns   = (const float*)d_in[1];
  const int*   perm = (const int*)d_in[2];
  const float* Uw   = (const float*)d_in[3];
  const float* Ub   = (const float*)d_in[4];
  const float* Ww   = (const float*)d_in[5];
  const float* Wb   = (const float*)d_in[6];
  const float* spbs = (const float*)d_in[7];
  const float* Pw   = (const float*)d_in[8];
  const float* Pb   = (const float*)d_in[9];
  const float* psw  = (const float*)d_in[10];
  const float* psb  = (const float*)d_in[11];
  const float* frsw = (const float*)d_in[12];
  const float* frsb = (const float*)d_in[13];
  const float* fvsw = (const float*)d_in[14];
  const float* fvsb = (const float*)d_in[15];
  const float* fbs  = (const float*)d_in[16];
  const float* fpjw = (const float*)d_in[17];
  const float* fpjb = (const float*)d_in[18];
  const float* brsw = (const float*)d_in[19];
  const float* brsb = (const float*)d_in[20];
  const float* bvsw = (const float*)d_in[21];
  const float* bvsb = (const float*)d_in[22];
  const float* bbs  = (const float*)d_in[23];
  const float* bpjw = (const float*)d_in[24];
  const float* bpjb = (const float*)d_in[25];
  const float* fp1w = (const float*)d_in[26];
  const float* fp1b = (const float*)d_in[27];
  const float* fp2w = (const float*)d_in[28];
  const float* fp2b = (const float*)d_in[29];
  const float* bp1w = (const float*)d_in[30];
  const float* bp1b = (const float*)d_in[31];
  const float* bp2w = (const float*)d_in[32];
  const float* bp2b = (const float*)d_in[33];

  char* ws = (char*)d_ws;
  _Float16* wfrag = (_Float16*)(ws + WS_WFRAG);
  _Float16* ufrag = (_Float16*)(ws + WS_UFRAG);
  int*      nbr   = (int*)     (ws + WS_NBR);
  float*    nbrw  = (float*)   (ws + WS_NBRW);
  float*    wproj = (float*)   (ws + WS_WPROJ);
  float*    mbias = (float*)   (ws + WS_MBIAS);
  float*    m4    = (float*)   (ws + WS_M4);
  _Float16* m16   = (_Float16*)(ws + WS_M16);
  float*    rws   = (float*)   (ws + WS_RWS);
  float*    out   = (float*)d_out;

  stg_prep_frag<<<576, 256, 0, stream>>>(Ww, frsw, fvsw, brsw, bvsw, Uw,
                                         wfrag, ufrag);
  stg_prep_graph<<<1, 256, 0, stream>>>(Ns, Pw, Pb, psw, psb, nbr, nbrw,
                                        wproj, mbias);
  stg_chain<<<D_ * T_ * (B_ / 16), 256, 0, stream>>>(
      x, perm, wfrag, ufrag, Ub, Wb, spbs, nbr, nbrw, wproj, m4);
  stg_reduce_m<<<(T_ * B_ * H_) / 256, 256, 0, stream>>>(m4, mbias, m16);
  stg_trnn<<<16, 256, 0, stream>>>(m16, wfrag, frsb, fvsb, fbs, fp1w,
                                   brsb, bvsb, bbs, bp1w, rws);
  stg_final<<<1, 128, 0, stream>>>(rws, fpjw, fpjb, fp1w, fp1b, fp2w, fp2b,
                                   bpjw, bpjb, bp1w, bp1b, bp2w, bp2b, out);
}
